// VectorQuantizerEMA_72060961292338
// MI455X (gfx1250) — matrix operand order repaired and checked by reading, not compiled, not run on hardware
//
#include <hip/hip_runtime.h>

typedef __attribute__((ext_vector_type(16))) _Float16 v16h;
typedef __attribute__((ext_vector_type(8)))  float    v8f;

#define VQ_H   4
#define VQ_K   4096
#define VQ_D   256
#define VQ_N   16384          // 32*512*1024 / (H*D) rows
#define VQ_KT  (VQ_K / 16)    // 256 k-tiles of 16 codes
#define VQ_DC  (VQ_D / 32)    // 8 d-chunks of 32
#define WPB    4              // waves per block
#define TILE_HALVES 4096      // halves per (hi or lo) B k-tile: 8 dchunks*32 lanes*16

// ---------------------------------------------------------------------------
// CDNA5 async global->LDS copy (ASYNCcnt) -- 16B per lane per issue.
// LDS byte address = low 32 bits of the generic shared pointer (ISA 10.2:
// generic LDS addr = SHARED_BASE aperture | LDS offset; hw uses addr[31:0]).
// ---------------------------------------------------------------------------
__device__ __forceinline__ void async_copy_b128(const _Float16* gsrc,
                                                const _Float16* ldst) {
  unsigned lds_addr = (unsigned)(uintptr_t)ldst;
  asm volatile("global_load_async_to_lds_b128 %0, %1, off"
               :: "v"(lds_addr), "v"(gsrc)
               : "memory");
}
__device__ __forceinline__ void wait_async0() {
  asm volatile("s_wait_asynccnt 0x0" ::: "memory");
}

// ---------------------------------------------------------------------------
// Kernel 1: per-code squared norm + f16 hi/lo weight split, pre-swizzled into
// WMMA B-fragment lane order: [h][ktile][dchunk][lane][16 halves].
// B (32x16 f16): lane = (k%16) + 16*((d%32)/16), halves j = d%16.
// ---------------------------------------------------------------------------
__global__ __launch_bounds__(256) void vq_prep(const float* __restrict__ w,
                                               _Float16* __restrict__ whi,
                                               _Float16* __restrict__ wlo,
                                               float* __restrict__ w2) {
  const int lane = threadIdx.x & 31;
  const int wave = threadIdx.x >> 5;
  const int row  = blockIdx.x * 8 + wave;        // row in [0, H*K)
  const int h    = row >> 12;                    // K = 4096
  const int k    = row & (VQ_K - 1);
  const int kt   = k >> 4;
  const int kin  = k & 15;
  const float* src = w + (size_t)row * VQ_D;

  float s = 0.f;
  for (int j = lane; j < VQ_D; j += 32) {
    float v = src[j];
    s += v * v;
    _Float16 hi = (_Float16)v;
    _Float16 lo = (_Float16)(v - (float)hi);
    int c     = j >> 5;                          // d-chunk
    int laneB = kin + (((j >> 4) & 1) << 4);     // +16 for upper half of chunk
    int jj    = j & 15;
    size_t off = ((((size_t)h * VQ_KT + kt) * VQ_DC + c) * 32 + laneB) * 16 + jj;
    whi[off] = hi;
    wlo[off] = lo;
  }
  for (int m = 16; m >= 1; m >>= 1) s += __shfl_xor(s, m, 32);
  if (lane == 0) w2[row] = s;
}

// ---------------------------------------------------------------------------
// Kernel 2: 4 waves per block, 16 rows each (64 rows/block). B tiles staged
// to LDS once per block via async global->LDS, double buffered; 24 WMMAs per
// wave per k-tile consume them. argmin of w2 - 2*x.w; gather; loss partial.
// ---------------------------------------------------------------------------
__global__ __launch_bounds__(128) void vq_main(const float* __restrict__ x,
                                               const float* __restrict__ w,
                                               const _Float16* __restrict__ whi,
                                               const _Float16* __restrict__ wlo,
                                               const float* __restrict__ w2,
                                               float* __restrict__ out,
                                               float* __restrict__ partial) {
  __shared__ __align__(16) float    xs[WPB][16 * VQ_D];            // 64 KB
  __shared__ __align__(16) _Float16 bbuf[2][2][TILE_HALVES];       // 32 KB
  __shared__ int   sidx[WPB][16];
  __shared__ float lpart[WPB];

  const int tid     = threadIdx.x;
  const int lane    = tid & 31;
  const int wv      = tid >> 5;
  const int h       = blockIdx.y;
  const int rowBase = blockIdx.x * (16 * WPB) + wv * 16;
  const int M       = lane & 15;
  const int kgrp    = lane >> 4;              // 0 or 1

  // Stage this wave's 16-row x tile (coalesced along d).
  for (int e = lane; e < 16 * VQ_D; e += 32) {
    int r = e >> 8, d = e & 255;
    xs[wv][e] = x[(size_t)(rowBase + r) * (VQ_H * VQ_D) + h * VQ_D + d];
  }

  const size_t hbase = (size_t)h * VQ_KT * TILE_HALVES;
  const _Float16* gHi = whi + hbase;
  const _Float16* gLo = wlo + hbase;

  // Each wave copies a 2KB slice of each of hi/lo per tile: 8x b128 issues.
  const int slice = wv * 1024 + lane * 8;     // halves
  auto issue_tile = [&](int kt, int nb) {
    const _Float16* th = gHi + (size_t)kt * TILE_HALVES;
    const _Float16* tl = gLo + (size_t)kt * TILE_HALVES;
#pragma unroll
    for (int i = 0; i < 4; ++i) {
      async_copy_b128(th + slice + i * 256, &bbuf[nb][0][slice + i * 256]);
      async_copy_b128(tl + slice + i * 256, &bbuf[nb][1][slice + i * 256]);
    }
  };

  __syncthreads();   // xs visible (per-wave use, but keeps waves aligned)

  // Build A fragments (16x32 f16, contiguous k = 16*kgrp + j), hi + lo, per d-chunk.
  v16h ahi[VQ_DC], alo[VQ_DC];
#pragma unroll
  for (int c = 0; c < VQ_DC; ++c) {
#pragma unroll
    for (int j = 0; j < 16; ++j) {
      int v  = j >> 1;
      int kk = kgrp * 16 + j;
      float val = xs[wv][M * VQ_D + c * 32 + kk];
      _Float16 hi = (_Float16)val;
      ahi[c][j] = hi;
      alo[c][j] = (_Float16)(val - (float)hi);
    }
  }

  float best[8];
  int   bidx[8];
#pragma unroll
  for (int r = 0; r < 8; ++r) { best[r] = 3.4e38f; bidx[r] = 0; }

  const float* w2h = w2 + h * VQ_K;

  // Preload tile 0.
  issue_tile(0, 0);
  wait_async0();
  __syncthreads();

  for (int kt = 0; kt < VQ_KT; ++kt) {
    const int cb = kt & 1;
    if (kt + 1 < VQ_KT) issue_tile(kt + 1, cb ^ 1);

    v8f acc = {};
#pragma unroll
    for (int c = 0; c < VQ_DC; ++c) {
      v16h bh = *(const v16h*)&bbuf[cb][0][c * 512 + lane * 16];
      v16h bl = *(const v16h*)&bbuf[cb][1][c * 512 + lane * 16];
      acc = __builtin_amdgcn_wmma_f32_16x16x32_f16(false, ahi[c], false, bh,
                                                   (short)0, acc, false, false);
      acc = __builtin_amdgcn_wmma_f32_16x16x32_f16(false, alo[c], false, bh,
                                                   (short)0, acc, false, false);
      acc = __builtin_amdgcn_wmma_f32_16x16x32_f16(false, ahi[c], false, bl,
                                                   (short)0, acc, false, false);
    }
    int   kidx = kt * 16 + (lane & 15);
    float w2v  = w2h[kidx];
#pragma unroll
    for (int r = 0; r < 8; ++r) {
      float dd = w2v - 2.f * acc[r];        // x^2 is constant per row: dropped
      if (dd < best[r]) { best[r] = dd; bidx[r] = kidx; }
    }

    wait_async0();        // my copies for next buffer done
    __syncthreads();      // everyone's copies done / reads of cur buffer done
  }

  // Cross-lane argmin within each 16-lane group (same M); ties -> lower index.
#pragma unroll
  for (int r = 0; r < 8; ++r) {
    float d0 = best[r];
    int   i0 = bidx[r];
    for (int m = 8; m >= 1; m >>= 1) {
      float od = __shfl_xor(d0, m, 32);
      int   oi = __shfl_xor(i0, m, 32);
      if (od < d0 || (od == d0 && oi < i0)) { d0 = od; i0 = oi; }
    }
    if ((lane & 15) == 0) sidx[wv][r + 8 * kgrp] = i0;
  }
  __syncthreads();

  // Gather fp32 codes to output (+1 for the loss scalar) and accumulate loss.
  float lsum = 0.f;
  float* qout = out + 1;
  for (int rr = 0; rr < 16; ++rr) {
    int kmin = sidx[wv][rr];
    const float4* src = (const float4*)(w + ((size_t)h * VQ_K + kmin) * VQ_D);
    float4* dst = (float4*)(qout + (size_t)(rowBase + rr) * (VQ_H * VQ_D) + h * VQ_D);
#pragma unroll
    for (int j4 = 0; j4 < 2; ++j4) {
      int j = lane + j4 * 32;
      float4 v = src[j];
      dst[j] = v;
      float a = v.x - xs[wv][rr * VQ_D + j * 4 + 0];
      float b = v.y - xs[wv][rr * VQ_D + j * 4 + 1];
      float c = v.z - xs[wv][rr * VQ_D + j * 4 + 2];
      float d = v.w - xs[wv][rr * VQ_D + j * 4 + 3];
      lsum += a * a + b * b + c * c + d * d;
    }
  }
  for (int m = 16; m >= 1; m >>= 1) lsum += __shfl_xor(lsum, m, 32);
  if (lane == 0) lpart[wv] = lsum;
  __syncthreads();
  if (tid == 0) {
    float t = 0.f;
#pragma unroll
    for (int i = 0; i < WPB; ++i) t += lpart[i];
    partial[blockIdx.y * gridDim.x + blockIdx.x] = t;
  }
}

// ---------------------------------------------------------------------------
// Kernel 3: deterministic reduction of block partials -> loss scalar.
// ---------------------------------------------------------------------------
__global__ __launch_bounds__(256) void vq_loss(const float* __restrict__ partial,
                                               float* __restrict__ out) {
  __shared__ float s[8];
  const int tid = threadIdx.x, lane = tid & 31, wave = tid >> 5;
  float v = 0.f;
  const int nPart = VQ_H * (VQ_N / (16 * WPB));
  for (int i = tid; i < nPart; i += 256) v += partial[i];
  for (int m = 16; m >= 1; m >>= 1) v += __shfl_xor(v, m, 32);
  if (lane == 0) s[wave] = v;
  __syncthreads();
  if (tid == 0) {
    float t = 0.f;
    for (int i = 0; i < 8; ++i) t += s[i];
    out[0] = 0.25f * t / 16777216.0f;   // COMMITMENT_COST * mean
  }
}

extern "C" void kernel_launch(void* const* d_in, const int* in_sizes, int n_in,
                              void* d_out, int out_size, void* d_ws, size_t ws_size,
                              hipStream_t stream) {
  const float* x = (const float*)d_in[0];   // (32,512,1024) fp32
  const float* w = (const float*)d_in[1];   // (4,4096,256)  fp32
  float* out = (float*)d_out;               // [loss, quantized...]

  char* ws = (char*)d_ws;
  const size_t halfBytes = (size_t)VQ_H * VQ_K * VQ_D * sizeof(_Float16); // 8 MB
  _Float16* whi = (_Float16*)ws;
  _Float16* wlo = (_Float16*)(ws + halfBytes);
  float* w2      = (float*)(ws + 2 * halfBytes);
  float* partial = (float*)(ws + 2 * halfBytes + (size_t)VQ_H * VQ_K * sizeof(float));

  vq_prep<<<dim3((VQ_H * VQ_K) / 8), 256, 0, stream>>>(w, whi, wlo, w2);
  vq_main<<<dim3(VQ_N / (16 * WPB), VQ_H), 32 * WPB, 0, stream>>>(x, w, whi, wlo, w2,
                                                                  out, partial);
  vq_loss<<<1, 256, 0, stream>>>(partial, out);
}
